// TripletLoss_23802708754519
// MI455X (gfx1250) — matrix operand order repaired and checked by reading, not compiled, not run on hardware
//
#include <hip/hip_runtime.h>
#include <hip/hip_bf16.h>

// ---------------------------------------------------------------------------
// Triplet loss (batch-hard), B=4096, D=256 fp32, 64 labels.
// Gram via v_wmma_f32_16x16x32_bf16. Block = 8 waves x 16-row bands (128 rows)
// sharing LDS-staged B column tiles (8x less L2 traffic than per-wave VMEM).
// Double-buffered LDS staging, 2 column tiles (16 WMMAs) per barrier phase.
// Hard-pos/neg mined on squared distances; merged across column-group blocks
// with order-invariant atomicMax/Min on monotone float->uint keys.
// ---------------------------------------------------------------------------

typedef __attribute__((ext_vector_type(16))) __bf16 v16bf;
typedef __attribute__((ext_vector_type(8)))  float  v8f;

#define BATCH    4096
#define DIM      256
#define NTILES   (BATCH / 16)     // 256 column tiles
#define ROWBLK   128              // rows per block (8 waves x 16)
#define COLGRP   4                // column groups (blocks in y)
#define TPG      (NTILES / COLGRP) // 64 tiles per group
#define MARGIN_F 0.3f
#define BIGF     3.0e38f
#define LSTRIDE  264              // 256 + 8 bf16 pad -> conflict-free ds_b128

union V16 { v16bf v; float4 f4[2]; };

// Order-preserving float <-> uint key (works for +-BIGF sentinels too)
__device__ __forceinline__ unsigned fenc(float f) {
    int i = __float_as_int(f);
    return (i >= 0) ? ((unsigned)i | 0x80000000u) : ~(unsigned)i;
}
__device__ __forceinline__ float fdec(unsigned u) {
    int i = (u & 0x80000000u) ? (int)(u & 0x7fffffffu) : (int)(~u);
    return __int_as_float(i);
}

// ---- Kernel 1: bf16 copy, row norms, init per-row min/max keys -------------
__launch_bounds__(256)
__global__ void tl_prep(const float* __restrict__ E,
                        __bf16* __restrict__ Ebf,
                        float* __restrict__ sq,
                        unsigned* __restrict__ posKey,
                        unsigned* __restrict__ negKey)
{
    const int row = blockIdx.x;      // 4096 blocks, one row each
    const int t   = threadIdx.x;     // 256 threads = one row
    float x = E[(size_t)row * DIM + t];
    Ebf[(size_t)row * DIM + t] = (__bf16)x;

    float s = x * x;
    for (int off = 16; off >= 1; off >>= 1)
        s += __shfl_xor(s, off, 32);

    __shared__ float wsum[8];
    if ((t & 31) == 0) wsum[t >> 5] = s;
    __syncthreads();
    if (t == 0) {
        float tot = 0.0f;
        #pragma unroll
        for (int w = 0; w < 8; ++w) tot += wsum[w];
        sq[row] = tot;
        posKey[row] = fenc(-BIGF);   // re-init every launch (graph replay safe)
        negKey[row] = fenc( BIGF);
    }
}

// ---- Kernel 2: WMMA Gram tiles, LDS-shared B, fused hard mining -----------
__launch_bounds__(256)
__global__ void tl_tiles(const __bf16* __restrict__ Ebf,
                         const float* __restrict__ sq,
                         const int*  __restrict__ labels,
                         unsigned* __restrict__ posKey,
                         unsigned* __restrict__ negKey)
{
    // double-buffered B panel: 2 x (32 rows x 264 bf16)  = 33 KB
    __shared__ __align__(16) __bf16 ldsB[2][32][LSTRIDE];

    const int tid  = threadIdx.x;
    const int wave = tid >> 5;        // 8 waves, each owns a 16-row band
    const int lane = tid & 31;
    const int half = lane >> 4;
    const int l15  = lane & 15;
    const int rowBase = blockIdx.x * ROWBLK + wave * 16;
    const int colGrp  = blockIdx.y;          // 0..3, 64 column tiles each

    // --- Preload A (16x256 of this wave's rows): 8 chunks of 16x32 bf16 ----
    V16 A[8];
    #pragma unroll
    for (int c = 0; c < 8; ++c) {
        const __bf16* p = Ebf + (size_t)(rowBase + l15) * DIM + 32 * c + 8 * half;
        A[c].f4[0] = *reinterpret_cast<const float4*>(p);
        A[c].f4[1] = *reinterpret_cast<const float4*>(p + 16);
    }

    float sqi[8]; int labi[8]; int rowi[8];
    #pragma unroll
    for (int v = 0; v < 8; ++v) {
        const int i = rowBase + v + 8 * half;
        rowi[v] = i;  sqi[v] = sq[i];  labi[v] = labels[i];
    }

    float posmax[8], negmin[8];   // mined on squared distances
    #pragma unroll
    for (int v = 0; v < 8; ++v) { posmax[v] = -BIGF; negmin[v] = BIGF; }

    // staging role: thread t copies 32 bf16 of row r (of the 32-row panel)
    const int sr = tid >> 3;          // 0..31 panel row
    const int ss = tid & 7;           // 0..7  segment (32 bf16 each)
    const int colBase0 = colGrp * TPG * 16;   // first column of this group

    // preload first panel into registers
    float4 st[4];
    {
        const float4* gp = reinterpret_cast<const float4*>(
            Ebf + (size_t)(colBase0 + sr) * DIM + ss * 32);
        #pragma unroll
        for (int q = 0; q < 4; ++q) st[q] = gp[q];
    }

    const int nIter = TPG / 2;        // 32 iterations, 2 tiles each
    for (int it = 0; it < nIter; ++it) {
        const int buf = it & 1;
        // store staged panel
        {
            float4* lp = reinterpret_cast<float4*>(&ldsB[buf][sr][ss * 32]);
            #pragma unroll
            for (int q = 0; q < 4; ++q) lp[q] = st[q];
        }
        __syncthreads();

        // issue next panel's global loads (latency covered by 16 WMMAs below)
        if (it + 1 < nIter) {
            const int nb = colBase0 + (it + 1) * 32;
            const float4* gp = reinterpret_cast<const float4*>(
                Ebf + (size_t)(nb + sr) * DIM + ss * 32);
            #pragma unroll
            for (int q = 0; q < 4; ++q) st[q] = gp[q];
        }

        // --- 2 column tiles from LDS, 2 independent WMMA chains ------------
        v8f acc0 = {0.f,0.f,0.f,0.f,0.f,0.f,0.f,0.f};
        v8f acc1 = {0.f,0.f,0.f,0.f,0.f,0.f,0.f,0.f};
        #pragma unroll
        for (int c = 0; c < 8; ++c) {
            // B layout: lane = N, lanes0-15 K 0..7,16..23, lanes16-31 K 8..15,24..31
            V16 B0, B1;
            const __bf16* p0 = &ldsB[buf][l15     ][32 * c + 8 * half];
            const __bf16* p1 = &ldsB[buf][16 + l15][32 * c + 8 * half];
            B0.f4[0] = *reinterpret_cast<const float4*>(p0);
            B0.f4[1] = *reinterpret_cast<const float4*>(p0 + 16);
            B1.f4[0] = *reinterpret_cast<const float4*>(p1);
            B1.f4[1] = *reinterpret_cast<const float4*>(p1 + 16);
            acc0 = __builtin_amdgcn_wmma_f32_16x16x32_bf16(
                       false, A[c].v, false, B0.v, (short)0, acc0, false, false);
            acc1 = __builtin_amdgcn_wmma_f32_16x16x32_bf16(
                       false, A[c].v, false, B1.v, (short)0, acc1, false, false);
        }

        // --- fused mining: element (v,lane): i = rowBase+v+8*half ----------
        const int cb = colBase0 + it * 32;
        {
            const int   j    = cb + l15;
            const float sqj  = sq[j];
            const int   labj = labels[j];
            #pragma unroll
            for (int v = 0; v < 8; ++v) {
                const float d2   = fmaf(-2.0f, acc0[v], sqi[v] + sqj);
                const bool  same = (labi[v] == labj);
                const bool  self = (rowi[v] == j);
                posmax[v] = (same && !self) ? fmaxf(posmax[v], d2) : posmax[v];
                negmin[v] = (!same)         ? fminf(negmin[v], d2) : negmin[v];
            }
        }
        {
            const int   j    = cb + 16 + l15;
            const float sqj  = sq[j];
            const int   labj = labels[j];
            #pragma unroll
            for (int v = 0; v < 8; ++v) {
                const float d2   = fmaf(-2.0f, acc1[v], sqi[v] + sqj);
                const bool  same = (labi[v] == labj);
                const bool  self = (rowi[v] == j);
                posmax[v] = (same && !self) ? fmaxf(posmax[v], d2) : posmax[v];
                negmin[v] = (!same)         ? fminf(negmin[v], d2) : negmin[v];
            }
        }
        __syncthreads();   // panel fully consumed before next store
    }

    // --- reduce over N within 16-lane halves, then global key atomics ------
    #pragma unroll
    for (int off = 8; off >= 1; off >>= 1) {
        #pragma unroll
        for (int v = 0; v < 8; ++v) {
            posmax[v] = fmaxf(posmax[v], __shfl_xor(posmax[v], off, 32));
            negmin[v] = fminf(negmin[v], __shfl_xor(negmin[v], off, 32));
        }
    }
    if (l15 == 0) {    // lanes 0 and 16 hold rows rowBase+v and rowBase+8+v
        #pragma unroll
        for (int v = 0; v < 8; ++v) {
            atomicMax(&posKey[rowi[v]], fenc(posmax[v]));
            atomicMin(&negKey[rowi[v]], fenc(negmin[v]));
        }
    }
}

// ---- Kernel 3: per-row loss + deterministic per-block partials ------------
__launch_bounds__(256)
__global__ void tl_loss(const unsigned* __restrict__ posKey,
                        const unsigned* __restrict__ negKey,
                        float* __restrict__ partial)   // [16 blocks][2]
{
    const int row = blockIdx.x * 256 + threadIdx.x;    // 16 blocks
    const float P2 = fdec(posKey[row]);
    const float N2 = fdec(negKey[row]);
    const bool valid = (P2 > -1.0e38f) && (N2 < 1.0e38f);
    float loss = valid ? fmaxf(sqrtf(fmaxf(P2, 0.0f)) - sqrtf(fmaxf(N2, 0.0f))
                               + MARGIN_F, 0.0f) : 0.0f;
    float cnt  = valid ? 1.0f : 0.0f;

    for (int off = 16; off >= 1; off >>= 1) {
        loss += __shfl_xor(loss, off, 32);
        cnt  += __shfl_xor(cnt,  off, 32);
    }
    __shared__ float ls[8], cs[8];
    if ((threadIdx.x & 31) == 0) { ls[threadIdx.x >> 5] = loss; cs[threadIdx.x >> 5] = cnt; }
    __syncthreads();
    if (threadIdx.x == 0) {
        float s = 0.0f, c = 0.0f;
        #pragma unroll
        for (int w = 0; w < 8; ++w) { s += ls[w]; c += cs[w]; }
        partial[blockIdx.x * 2 + 0] = s;
        partial[blockIdx.x * 2 + 1] = c;
    }
}

// ---- Kernel 4: deterministic finalize -------------------------------------
__global__ void tl_finalize(const float* __restrict__ partial, float* __restrict__ out)
{
    if (blockIdx.x == 0 && threadIdx.x == 0) {
        float s = 0.0f, c = 0.0f;
        for (int b = 0; b < 16; ++b) { s += partial[b * 2]; c += partial[b * 2 + 1]; }
        out[0] = (c > 0.0f) ? s / fmaxf(c, 1.0f) : 0.0f;
    }
}

extern "C" void kernel_launch(void* const* d_in, const int* in_sizes, int n_in,
                              void* d_out, int out_size, void* d_ws, size_t ws_size,
                              hipStream_t stream)
{
    const float* E      = (const float*)d_in[0];   // [4096, 256] fp32
    const int*   labels = (const int*)d_in[1];     // [4096] int
    float*       out    = (float*)d_out;           // scalar

    char*     ws      = (char*)d_ws;
    __bf16*   Ebf     = (__bf16*)ws;                                     // 2 MB
    float*    sq      = (float*)(ws + (size_t)BATCH * DIM * sizeof(__bf16));
    unsigned* posKey  = (unsigned*)((char*)sq + BATCH * sizeof(float));
    unsigned* negKey  = posKey + BATCH;
    float*    partial = (float*)(negKey + BATCH);                        // 16*2 floats

    tl_prep    <<<BATCH, 256, 0, stream>>>(E, Ebf, sq, posKey, negKey);
    tl_tiles   <<<dim3(BATCH / ROWBLK, COLGRP), 256, 0, stream>>>(Ebf, sq, labels,
                                                                  posKey, negKey);
    tl_loss    <<<BATCH / 256, 256, 0, stream>>>(posKey, negKey, partial);
    tl_finalize<<<1, 32, 0, stream>>>(partial, out);
}
